// CustomGPT_23862838296729
// MI455X (gfx1250) — matrix operand order repaired and checked by reading, not compiled, not run on hardware
//
#include <hip/hip_runtime.h>
#include <cstddef>
#include <cstdint>

// ---------------- types ----------------
typedef __bf16 bf16_t;
typedef __attribute__((ext_vector_type(16))) __bf16 v16bf;
typedef __attribute__((ext_vector_type(8)))  __bf16 v8bf;
typedef __attribute__((ext_vector_type(8)))  float  v8f;

#define DM 512      // model dim
#define TT 1024     // seq len
#define NL 6        // layers
#define NH 8        // heads
#define HS 64       // head size
#define NB 2        // batch
#define FFD 2048    // ffn dim
#define NV 50257    // vocab

__device__ __forceinline__ v16bf combine16(v8bf lo, v8bf hi) {
  union { v16bf v; v8bf h[2]; } u;
  u.h[0] = lo; u.h[1] = hi;
  return u.v;
}
__device__ __forceinline__ v8bf ld8(const bf16_t* p) { return *(const v8bf*)p; }

__device__ __forceinline__ v8f wmma_bf16(v16bf a, v16bf b, v8f c) {
  // D = A(16x32 bf16) * B(32x16 bf16) + C(16x16 f32)
  return __builtin_amdgcn_wmma_f32_16x16x32_bf16(false, a, false, b, (short)0, c,
                                                 false, false);
}

// async global -> LDS copy of 16 bytes (ASYNCcnt-tracked, no VGPR data path)
__device__ __forceinline__ void async_cp16(const void* gptr, void* lptr) {
  unsigned lds_off = (unsigned)(uintptr_t)lptr;  // low 32 bits = LDS byte offset
  asm volatile("global_load_async_to_lds_b128 %0, %1, off"
               :: "v"(lds_off), "v"(gptr) : "memory");
}
__device__ __forceinline__ void async_wait0() {
  asm volatile("s_wait_asynccnt 0x0" ::: "memory");
}

// ------- weight cast + transpose: src[K,N] f32 -> dst[N,K] bf16 (64x64 tiles) ---
__global__ __launch_bounds__(256) void cvt_transpose(
    const float* __restrict__ src, bf16_t* __restrict__ dst, int K, int N) {
  __shared__ bf16_t tile[64][65];
  const int k0 = blockIdx.y * 64, n0 = blockIdx.x * 64;
  const int tid = threadIdx.x;
#pragma unroll
  for (int p = 0; p < 16; ++p) {
    int idx = tid + p * 256;
    int kk = idx >> 6, nn = idx & 63;      // coalesced read over n
    float v = 0.0f;
    if (n0 + nn < N) v = src[(size_t)(k0 + kk) * N + n0 + nn];
    tile[nn][kk] = (bf16_t)v;
  }
  __syncthreads();
#pragma unroll
  for (int p = 0; p < 16; ++p) {
    int idx = tid + p * 256;
    int nn = idx >> 6, kk = idx & 63;      // coalesced write over k
    if (n0 + nn < N) dst[(size_t)(n0 + nn) * K + k0 + kk] = tile[nn][kk];
  }
}

// ---------------- embedding ----------------
__global__ void embed_add(const int* __restrict__ idx, const float* __restrict__ tok,
                          const float* __restrict__ pos, float* __restrict__ x) {
  int i = blockIdx.x * 256 + threadIdx.x;
  if (i >= NB * TT * DM) return;
  int d = i & (DM - 1);
  int bt = i >> 9;             // DM == 512
  int t = bt & (TT - 1);
  x[i] = tok[(size_t)idx[bt] * DM + d] + pos[(size_t)t * DM + d];
}

// ---------------- layernorm (f32 in, bf16 out) ----------------
__global__ __launch_bounds__(256) void layernorm_to_bf16(
    const float* __restrict__ x, const float* __restrict__ gamma,
    const float* __restrict__ beta, bf16_t* __restrict__ out) {
  __shared__ float red[256];
  const int row = blockIdx.x, tid = threadIdx.x;
  const float* xr = x + (size_t)row * DM;
  float v0 = xr[tid], v1 = xr[tid + 256];
  red[tid] = v0 + v1;
  __syncthreads();
  for (int s = 128; s > 0; s >>= 1) {
    if (tid < s) red[tid] += red[tid + s];
    __syncthreads();
  }
  float mean = red[0] * (1.0f / DM);
  __syncthreads();
  float d0 = v0 - mean, d1 = v1 - mean;
  red[tid] = d0 * d0 + d1 * d1;
  __syncthreads();
  for (int s = 128; s > 0; s >>= 1) {
    if (tid < s) red[tid] += red[tid + s];
    __syncthreads();
  }
  float rstd = rsqrtf(red[0] * (1.0f / DM) + 1e-5f);
  bf16_t* orow = out + (size_t)row * DM;
  orow[tid]       = (bf16_t)(d0 * rstd * gamma[tid] + beta[tid]);
  orow[tid + 256] = (bf16_t)(d1 * rstd * gamma[tid + 256] + beta[tid + 256]);
}

// ------------- bf16 WMMA GEMM with async double-buffered LDS tiles -------------
// A: [M,K] bf16 row-major, BT: [N,K] bf16 row-major (weights pre-transposed).
// Block tile 128x128, 8 waves (4x2), wave tile 32x64 = 2x4 WMMA frags, BK=32.
// Tiles are copied global->LDS by the async engine (ASYNCcnt) while the waves
// run WMMAs on the other buffer.
__global__ __launch_bounds__(256) void gemm_bf16(
    const bf16_t* __restrict__ A, int lda, const bf16_t* __restrict__ BT, int ldb,
    const float* __restrict__ bias, const float* __restrict__ resid,
    float* __restrict__ outF, bf16_t* __restrict__ outB, bf16_t* __restrict__ outBT,
    int M, int N, int K, int act_gelu) {
  __shared__ bf16_t As[2][128][40];   // [buf][m][k], padded rows
  __shared__ bf16_t Bs[2][128][40];   // [buf][n][k], padded rows
  const int tid = threadIdx.x;
  const int lane = tid & 31, wid = tid >> 5;
  const int lanen = lane & 15, half = lane >> 4;
  const int wr = wid >> 1, wc = wid & 1;           // 4x2 wave grid
  const int m0 = blockIdx.y * 128, n0 = blockIdx.x * 128;

  // this thread's two 16B chunks: (row, kc) ownership is fixed
  const int row0 = tid >> 2,          kc0 = (tid & 3) * 8;
  const int row1 = (tid + 256) >> 2,  kc1 = ((tid + 256) & 3) * 8;

  // zero OOB B rows once (async copies below are predicated off for them)
  if (n0 + 128 > N) {
    uint4 z{};
    if (n0 + row0 >= N) { *(uint4*)&Bs[0][row0][kc0] = z; *(uint4*)&Bs[1][row0][kc0] = z; }
    if (n0 + row1 >= N) { *(uint4*)&Bs[0][row1][kc1] = z; *(uint4*)&Bs[1][row1][kc1] = z; }
  }

  auto issue_tile = [&](int k0, int buf) {
    async_cp16(A + (size_t)(m0 + row0) * lda + k0 + kc0, &As[buf][row0][kc0]);
    async_cp16(A + (size_t)(m0 + row1) * lda + k0 + kc1, &As[buf][row1][kc1]);
    if (n0 + row0 < N) async_cp16(BT + (size_t)(n0 + row0) * ldb + k0 + kc0,
                                  &Bs[buf][row0][kc0]);
    if (n0 + row1 < N) async_cp16(BT + (size_t)(n0 + row1) * ldb + k0 + kc1,
                                  &Bs[buf][row1][kc1]);
  };

  v8f acc[2][4] = {{{}, {}, {}, {}}, {{}, {}, {}, {}}};

  const int nk = K >> 5;          // K multiple of 32
  issue_tile(0, 0);
  for (int it = 0; it < nk; ++it) {
    const int buf = it & 1;
    async_wait0();                // tile `it` landed in LDS
    __syncthreads();              // visible to all waves; prev buf free to overwrite
    if (it + 1 < nk) issue_tile((it + 1) * 32, buf ^ 1);  // overlap copy w/ compute

    v16bf af[2], bfr[4];
#pragma unroll
    for (int i = 0; i < 2; ++i) {
      const bf16_t* ap = &As[buf][wr * 32 + i * 16 + lanen][half * 8];
      af[i] = combine16(ld8(ap), ld8(ap + 16));      // A-frag K chunks {h*8, h*8+16}
    }
#pragma unroll
    for (int j = 0; j < 4; ++j) {
      const bf16_t* bp = &Bs[buf][wc * 64 + j * 16 + lanen][half * 8];
      bfr[j] = combine16(ld8(bp), ld8(bp + 16));     // B-frag K chunks {h*8, h*8+16}
    }
#pragma unroll
    for (int i = 0; i < 2; ++i)
#pragma unroll
      for (int j = 0; j < 4; ++j)
        acc[i][j] = wmma_bf16(af[i], bfr[j], acc[i][j]);
  }

  // epilogue: bias / gelu / residual / stores  (C layout: row=e+8*half, col=lanen)
#pragma unroll
  for (int i = 0; i < 2; ++i) {
#pragma unroll
    for (int j = 0; j < 4; ++j) {
      int n = n0 + wc * 64 + j * 16 + lanen;
      if (n >= N) continue;
      float bsv = bias[n];
#pragma unroll
      for (int e = 0; e < 8; ++e) {
        int m = m0 + wr * 32 + i * 16 + e + half * 8;
        float v = acc[i][j][e] + bsv;
        if (act_gelu) v = 0.5f * v * (1.0f + erff(v * 0.70710678118654752f));
        size_t o = (size_t)m * N + n;
        if (resid) v += resid[o];
        if (outF) outF[o] = v;
        if (outB) outB[o] = (bf16_t)v;
        if (outBT) {  // per-batch transposed: [b][n][t], N is feature dim
          int bb = m >> 10;          // m / TT
          int t2 = m & (TT - 1);
          outBT[((size_t)bb * N + n) * TT + t2] = (bf16_t)v;
        }
      }
    }
  }
}

// ---------------- flash attention: one wave per (b, h, 16-query tile) ----------
__global__ __launch_bounds__(128) void attn_flash(
    const bf16_t* __restrict__ q,    // [B,T,512]
    const bf16_t* __restrict__ kk,   // [B,T,512]
    const bf16_t* __restrict__ vT,   // [B,512,T] (transposed by V-proj GEMM)
    bf16_t* __restrict__ att) {      // [B,T,512]
  __shared__ bf16_t pbuf[4][16][32];  // per-wave P tile (C-layout -> A-layout bounce)
  const int lane = threadIdx.x & 31, wid = threadIdx.x >> 5;
  const int lanen = lane & 15, half = lane >> 4;
  const int g = blockIdx.x * 4 + wid;
  const int qt = g & 63;             // T/16 = 64 tiles
  const int hh = (g >> 6) & 7;
  const int b  = g >> 9;
  const int q0 = qt * 16;
  const float scale = 0.125f;        // 1/sqrt(64)

  // Q A-fragments for HS=64 (two K=32 chunks)
  const bf16_t* qp = q + ((size_t)(b * TT + q0 + lanen)) * DM + hh * HS;
  v16bf qa0 = combine16(ld8(qp + half * 8), ld8(qp + half * 8 + 16));
  v16bf qa1 = combine16(ld8(qp + 32 + half * 8), ld8(qp + 32 + half * 8 + 16));

  v8f o0 = {}, o1 = {}, o2 = {}, o3 = {};
  float mrow[8], lrow[8];
#pragma unroll
  for (int e = 0; e < 8; ++e) { mrow[e] = -1e30f; lrow[e] = 0.0f; }

  const int kend = q0 + 16;                 // causal: keys <= q0+15
  for (int kc = 0; kc < kend; kc += 32) {
    float sv[2][8];
#pragma unroll
    for (int t = 0; t < 2; ++t) {
      const int kb16 = kc + t * 16;
      // K B-fragment: col = key, reduction = hs (chunks at half*8, half*8+16)
      const bf16_t* kp =
          kk + ((size_t)(b * TT + kb16 + lanen)) * DM + hh * HS + half * 8;
      v16bf kf0 = combine16(ld8(kp), ld8(kp + 16));
      v16bf kf1 = combine16(ld8(kp + 32), ld8(kp + 48));
      v8f s = {};
      s = wmma_bf16(qa0, kf0, s);
      s = wmma_bf16(qa1, kf1, s);
#pragma unroll
      for (int e = 0; e < 8; ++e) {
        int qrow = q0 + e + half * 8;
        int key = kb16 + lanen;
        sv[t][e] = (key <= qrow) ? s[e] * scale : -1e30f;
      }
    }
    // online softmax over this 32-key chunk (row reductions across 16 lanes)
#pragma unroll
    for (int e = 0; e < 8; ++e) {
      float tm = fmaxf(sv[0][e], sv[1][e]);
      tm = fmaxf(tm, __shfl_xor(tm, 1, 32));
      tm = fmaxf(tm, __shfl_xor(tm, 2, 32));
      tm = fmaxf(tm, __shfl_xor(tm, 4, 32));
      tm = fmaxf(tm, __shfl_xor(tm, 8, 32));
      float nm = fmaxf(mrow[e], tm);
      float corr = __expf(mrow[e] - nm);
      float p0 = __expf(sv[0][e] - nm);
      float p1 = __expf(sv[1][e] - nm);
      float rs = p0 + p1;
      rs += __shfl_xor(rs, 1, 32);
      rs += __shfl_xor(rs, 2, 32);
      rs += __shfl_xor(rs, 4, 32);
      rs += __shfl_xor(rs, 8, 32);
      lrow[e] = lrow[e] * corr + rs;
      mrow[e] = nm;
      o0[e] *= corr; o1[e] *= corr; o2[e] *= corr; o3[e] *= corr;
      int prow = e + half * 8;
      pbuf[wid][prow][lanen]      = (bf16_t)p0;
      pbuf[wid][prow][lanen + 16] = (bf16_t)p1;
    }
    // re-read P in A-fragment layout (same wave; compiler orders LDS deps)
    const bf16_t* pp = &pbuf[wid][lanen][half * 8];
    v16bf pa = combine16(ld8(pp), ld8(pp + 16));
    // P(16x32) * V(32x16 per hs tile); vT gives contiguous key-runs per feature
    const bf16_t* vp =
        vT + ((size_t)(b * DM + hh * HS + lanen)) * TT + kc + half * 8;
    o0 = wmma_bf16(pa, combine16(ld8(vp + 0 * 16 * TT), ld8(vp + 0 * 16 * TT + 16)), o0);
    o1 = wmma_bf16(pa, combine16(ld8(vp + 1 * 16 * TT), ld8(vp + 1 * 16 * TT + 16)), o1);
    o2 = wmma_bf16(pa, combine16(ld8(vp + 2 * 16 * TT), ld8(vp + 2 * 16 * TT + 16)), o2);
    o3 = wmma_bf16(pa, combine16(ld8(vp + 3 * 16 * TT), ld8(vp + 3 * 16 * TT + 16)), o3);
  }

  bf16_t* op = att + ((size_t)(b * TT + q0)) * DM + hh * HS;
#pragma unroll
  for (int e = 0; e < 8; ++e) {
    float inv = 1.0f / lrow[e];
    int rr = e + half * 8;
    op[(size_t)rr * DM +  0 + lanen] = (bf16_t)(o0[e] * inv);
    op[(size_t)rr * DM + 16 + lanen] = (bf16_t)(o1[e] * inv);
    op[(size_t)rr * DM + 32 + lanen] = (bf16_t)(o2[e] * inv);
    op[(size_t)rr * DM + 48 + lanen] = (bf16_t)(o3[e] * inv);
  }
}

// ---------------- host ----------------
extern "C" void kernel_launch(void* const* d_in, const int* in_sizes, int n_in,
                              void* d_out, int out_size, void* d_ws, size_t ws_size,
                              hipStream_t stream) {
  (void)in_sizes; (void)n_in; (void)out_size; (void)ws_size;
  const int*   idx  = (const int*)d_in[0];
  const float* tok  = (const float*)d_in[1];
  const float* pos  = (const float*)d_in[2];
  const float* ln1g = (const float*)d_in[3];
  const float* ln1b = (const float*)d_in[4];
  const float* Wq   = (const float*)d_in[5];
  const float* bq   = (const float*)d_in[6];
  const float* Wk   = (const float*)d_in[7];
  const float* bk   = (const float*)d_in[8];
  const float* Wv   = (const float*)d_in[9];
  const float* bv   = (const float*)d_in[10];
  const float* Wo   = (const float*)d_in[11];
  const float* bo   = (const float*)d_in[12];
  const float* ln2g = (const float*)d_in[13];
  const float* ln2b = (const float*)d_in[14];
  const float* W1   = (const float*)d_in[15];
  const float* b1   = (const float*)d_in[16];
  const float* W2   = (const float*)d_in[17];
  const float* b2   = (const float*)d_in[18];
  const float* lnfg = (const float*)d_in[19];
  const float* lnfb = (const float*)d_in[20];
  const float* Wlm  = (const float*)d_in[21];
  const float* blm  = (const float*)d_in[22];

  char* ws = (char*)d_ws;
  size_t off = 0;
  auto carve = [&](size_t bytes) -> char* {
    char* p = ws + off;
    off += (bytes + 255) & ~(size_t)255;
    return p;
  };
  const int M = NB * TT;  // 2048 rows

  // pre-transposed bf16 weights: [N,K] per layer
  bf16_t* wq16  = (bf16_t*)carve((size_t)NL * DM * DM * 2);
  bf16_t* wk16  = (bf16_t*)carve((size_t)NL * DM * DM * 2);
  bf16_t* wv16  = (bf16_t*)carve((size_t)NL * DM * DM * 2);
  bf16_t* wo16  = (bf16_t*)carve((size_t)NL * DM * DM * 2);
  bf16_t* w116  = (bf16_t*)carve((size_t)NL * DM * FFD * 2);
  bf16_t* w216  = (bf16_t*)carve((size_t)NL * FFD * DM * 2);
  bf16_t* wlm16 = (bf16_t*)carve((size_t)DM * NV * 2);
  float*  xbuf  = (float*)carve((size_t)M * DM * 4);
  bf16_t* hbuf  = (bf16_t*)carve((size_t)M * DM * 2);
  bf16_t* qbuf  = (bf16_t*)carve((size_t)M * DM * 2);
  bf16_t* kbuf  = (bf16_t*)carve((size_t)M * DM * 2);
  bf16_t* vTbuf = (bf16_t*)carve((size_t)NB * DM * TT * 2);
  bf16_t* attb  = (bf16_t*)carve((size_t)M * DM * 2);
  bf16_t* ffbuf = (bf16_t*)carve((size_t)M * FFD * 2);

  auto cvtT = [&](const float* s, bf16_t* dp, int K, int N) {
    dim3 grid((N + 63) / 64, K / 64);
    cvt_transpose<<<grid, 256, 0, stream>>>(s, dp, K, N);
  };
  auto gemm = [&](const bf16_t* A, int lda, const bf16_t* Bp, int ldb,
                  const float* bias, const float* resid, float* oF, bf16_t* oB,
                  bf16_t* oBT, int Mr, int Nr, int Kr, int act) {
    dim3 grid((Nr + 127) / 128, Mr / 128);
    gemm_bf16<<<grid, dim3(256), 0, stream>>>(A, lda, Bp, ldb, bias, resid, oF, oB,
                                              oBT, Mr, Nr, Kr, act);
  };

  // weights -> transposed bf16 (deterministic each call)
  for (int l = 0; l < NL; ++l) {
    cvtT(Wq + (size_t)l * DM * DM, wq16 + (size_t)l * DM * DM, DM, DM);
    cvtT(Wk + (size_t)l * DM * DM, wk16 + (size_t)l * DM * DM, DM, DM);
    cvtT(Wv + (size_t)l * DM * DM, wv16 + (size_t)l * DM * DM, DM, DM);
    cvtT(Wo + (size_t)l * DM * DM, wo16 + (size_t)l * DM * DM, DM, DM);
    cvtT(W1 + (size_t)l * DM * FFD, w116 + (size_t)l * DM * FFD, DM, FFD);
    cvtT(W2 + (size_t)l * FFD * DM, w216 + (size_t)l * FFD * DM, FFD, DM);
  }
  cvtT(Wlm, wlm16, DM, NV);

  embed_add<<<(NB * TT * DM + 255) / 256, 256, 0, stream>>>(idx, tok, pos, xbuf);

  for (int l = 0; l < NL; ++l) {
    layernorm_to_bf16<<<M, 256, 0, stream>>>(xbuf, ln1g + l * DM, ln1b + l * DM, hbuf);
    gemm(hbuf, DM, wq16 + (size_t)l * DM * DM, DM, bq + l * DM, nullptr, nullptr,
         qbuf, nullptr, M, DM, DM, 0);
    gemm(hbuf, DM, wk16 + (size_t)l * DM * DM, DM, bk + l * DM, nullptr, nullptr,
         kbuf, nullptr, M, DM, DM, 0);
    gemm(hbuf, DM, wv16 + (size_t)l * DM * DM, DM, bv + l * DM, nullptr, nullptr,
         nullptr, vTbuf, M, DM, DM, 0);  // V stored transposed for attention
    attn_flash<<<256, 128, 0, stream>>>(qbuf, kbuf, vTbuf, attb);
    gemm(attb, DM, wo16 + (size_t)l * DM * DM, DM, bo + l * DM, xbuf, xbuf, nullptr,
         nullptr, M, DM, DM, 0);  // x += att @ Wo + bo
    layernorm_to_bf16<<<M, 256, 0, stream>>>(xbuf, ln2g + l * DM, ln2b + l * DM, hbuf);
    gemm(hbuf, DM, w116 + (size_t)l * DM * FFD, DM, b1 + l * FFD, nullptr, nullptr,
         ffbuf, nullptr, M, FFD, DM, 1);  // GELU; W1^T is [FFD, DM] so ldb = DM
    gemm(ffbuf, FFD, w216 + (size_t)l * FFD * DM, FFD, b2 + l * DM, xbuf, xbuf,
         nullptr, nullptr, M, DM, FFD, 0);  // x += ff @ W2 + b2; W2^T ldb = FFD
  }
  layernorm_to_bf16<<<M, 256, 0, stream>>>(xbuf, lnfg, lnfb, hbuf);
  gemm(hbuf, DM, wlm16, DM, blm, nullptr, (float*)d_out, nullptr, nullptr, M, NV,
       DM, 0);  // logits; Wlm^T is [NV, DM] so ldb = DM
}
